// MultiHeadedAttention_2396591751196
// MI455X (gfx1250) — compile-verified, matrix operand order checked by reading, not run on hardware
//
#include <hip/hip_runtime.h>
#include <hip/hip_bf16.h>
#include <stdint.h>

// ---------------------------------------------------------------------------
// Multi-head attention forward for gfx1250 (MI455X), wave32 + WMMA f16.
//   B=2, S=2048, D=1024, H=16, HD=64
// Pipeline:
//   1) cvt_f32_f16x4: x, Wq, Wk, Wv -> f16 copies in workspace (vectorized)
//   2) qkv_kernel   : WMMA GEMM C = x @ W^T + b; double-buffered tile regs
//                     pinned with sched_barrier; block waves share W rows
//                     (WGP$ reuse). Q pre-scaled by 1/sqrt(HD).
//   3) attn_kernel  : fused flash-attention; scores computed transposed so the
//                     softmaxed P tile is already in WMMA A-matrix layout;
//                     K/V tile loads software-pipelined around the WMMAs.
// ---------------------------------------------------------------------------

#define MODEL_DIM 1024
#define NUM_HEADS 16
#define HEAD_DIM  64
#define BATCH     2
#define SEQ       2048

#if __has_builtin(__builtin_amdgcn_sched_barrier)
#define SCHED_FENCE() __builtin_amdgcn_sched_barrier(0)
#else
#define SCHED_FENCE()
#endif

typedef __attribute__((ext_vector_type(16))) _Float16 v16h;
typedef __attribute__((ext_vector_type(8)))  _Float16 v8h;
typedef __attribute__((ext_vector_type(4)))  _Float16 v4h;
typedef __attribute__((ext_vector_type(8)))  float    v8f;
typedef __attribute__((ext_vector_type(4)))  float    v4f;

static __device__ __forceinline__ v8h ld8(const _Float16* p) {
    return *(const v8h*)p;
}
static __device__ __forceinline__ v16h cat16(v8h lo, v8h hi) {
    v16h r;
#pragma unroll
    for (int i = 0; i < 8; ++i) { r[i] = lo[i]; r[8 + i] = hi[i]; }
    return r;
}
// load one WMMA 16-wide K-slice for this lane: {base[0..7], base[16..23]}
static __device__ __forceinline__ v16h ldtile(const _Float16* p) {
    return cat16(ld8(p), ld8(p + 16));
}
static __device__ __forceinline__ v8f wmma_f16(v16h a, v16h b, v8f c) {
    return __builtin_amdgcn_wmma_f32_16x16x32_f16(
        /*neg_a=*/false, a, /*neg_b=*/false, b,
        /*c_mod=*/(short)0, c, /*reuse_a=*/false, /*reuse_b=*/false);
}

// ---------------------------------------------------------------------------
// 1) fp32 -> f16 conversion, 4 elements per thread
// ---------------------------------------------------------------------------
__global__ void cvt_f32_f16x4(const float* __restrict__ in,
                              _Float16* __restrict__ out, int n4) {
    int i = blockIdx.x * blockDim.x + threadIdx.x;
    if (i < n4) {
        v4f v = ((const v4f*)in)[i];
        v4h o;
#pragma unroll
        for (int j = 0; j < 4; ++j) o[j] = (_Float16)v[j];
        ((v4h*)out)[i] = o;
    }
}

// ---------------------------------------------------------------------------
// 2) QKV projection:  C[m,n] = sum_k x[m,k] * W[n,k] + bias[n]
//    One wave per 16x64 output strip. Job decomposition puts the 8 waves of a
//    block on the SAME n-strip (identical W rows -> WGP$ hits) and adjacent
//    m-tiles (contiguous x streaming). blockIdx.y selects Q/K/V.
// ---------------------------------------------------------------------------
__global__ void qkv_kernel(const _Float16* __restrict__ xh,
                           const _Float16* __restrict__ Wqh,
                           const _Float16* __restrict__ Wkh,
                           const _Float16* __restrict__ Wvh,
                           const float* __restrict__ bq,
                           const float* __restrict__ bk,
                           const float* __restrict__ bv,
                           _Float16* __restrict__ Qo,
                           _Float16* __restrict__ Ko,
                           _Float16* __restrict__ Vo) {
    const int lane = threadIdx.x & 31;
    const int wave = threadIdx.x >> 5;
    const int job  = blockIdx.x * (blockDim.x >> 5) + wave;   // 0..4095
    const int mt   = job & 255;       // 256 row-tiles of 16 (fast index)
    const int nt4  = job >> 8;        // 16 col-strips of 64  (shared in block)
    const int mat  = blockIdx.y;      // 0=Q 1=K 2=V

    const _Float16* W    = (mat == 0) ? Wqh : (mat == 1) ? Wkh : Wvh;
    const float*    bias = (mat == 0) ? bq  : (mat == 1) ? bk  : bv;

    const int hl  = lane >> 4;        // lane half selects the K-split
    const int l16 = lane & 15;
    const int n0  = nt4 * 64;

    const _Float16* arow  = xh + (size_t)(mt * 16 + l16) * MODEL_DIM + 8 * hl;
    const _Float16* wrow0 = W + (size_t)(n0 + 0 * 16 + l16) * MODEL_DIM + 8 * hl;
    const _Float16* wrow1 = W + (size_t)(n0 + 1 * 16 + l16) * MODEL_DIM + 8 * hl;
    const _Float16* wrow2 = W + (size_t)(n0 + 2 * 16 + l16) * MODEL_DIM + 8 * hl;
    const _Float16* wrow3 = W + (size_t)(n0 + 3 * 16 + l16) * MODEL_DIM + 8 * hl;

    v8f c0 = {}, c1 = {}, c2 = {}, c3 = {};

    // prologue: tiles for k = 0
    v16h a_c  = ldtile(arow);
    v16h b0_c = ldtile(wrow0);
    v16h b1_c = ldtile(wrow1);
    v16h b2_c = ldtile(wrow2);
    v16h b3_c = ldtile(wrow3);

#pragma unroll 4
    for (int k = 0; k < MODEL_DIM - 32; k += 32) {
        const int kn = k + 32;
        // issue next-step loads (kept above the WMMAs by the sched fence so
        // they are in flight while this step's WMMAs execute)
        v16h a_n  = ldtile(arow  + kn);
        v16h b0_n = ldtile(wrow0 + kn);
        v16h b1_n = ldtile(wrow1 + kn);
        v16h b2_n = ldtile(wrow2 + kn);
        v16h b3_n = ldtile(wrow3 + kn);

        SCHED_FENCE();   // pin loads(k+1) above wmmas(k)

        c0 = wmma_f16(a_c, b0_c, c0);
        c1 = wmma_f16(a_c, b1_c, c1);
        c2 = wmma_f16(a_c, b2_c, c2);
        c3 = wmma_f16(a_c, b3_c, c3);

        SCHED_FENCE();   // keep the rotate after the wmmas

        a_c = a_n; b0_c = b0_n; b1_c = b1_n; b2_c = b2_n; b3_c = b3_n;
    }
    // epilogue: last k-step
    c0 = wmma_f16(a_c, b0_c, c0);
    c1 = wmma_f16(a_c, b1_c, c1);
    c2 = wmma_f16(a_c, b2_c, c2);
    c3 = wmma_f16(a_c, b3_c, c3);

    // Fold attention score scale 1/sqrt(HD) into Q (values AND bias).
    const float oscale = (mat == 0) ? 0.125f : 1.0f;

    // C layout: lane = n (col), VGPR r = row (r + 8*lanehalf). Bias, store f16.
#pragma unroll
    for (int j = 0; j < 4; ++j) {
        v8f c = (j == 0) ? c0 : (j == 1) ? c1 : (j == 2) ? c2 : c3;
        const int n  = n0 + j * 16 + l16;
        const float bn = bias[n];
        const int h = n >> 6, d = n & 63;
#pragma unroll
        for (int r = 0; r < 8; ++r) {
            const int mg = mt * 16 + r + 8 * hl;     // global row in [0,4096)
            const int b  = mg >> 11;                 // batch
            const int s  = mg & 2047;                // seq pos
            const _Float16 val = (_Float16)((c[r] + bn) * oscale);
            if (mat == 2) {
                // V transposed: [b,h,d,s]
                Vo[(((size_t)(b * NUM_HEADS + h) * HEAD_DIM + d) * SEQ) + s] = val;
            } else {
                _Float16* dst = (mat == 0) ? Qo : Ko;  // [b,h,s,d]
                dst[(((size_t)(b * NUM_HEADS + h) * SEQ + s) * HEAD_DIM) + d] = val;
            }
        }
    }
}

// ---------------------------------------------------------------------------
// 3) Fused attention. One wave per (b, h, 16-query tile); online softmax over
//    SEQ keys in chunks of 32. Scores computed transposed (M=key, N=query) so
//    the exp'd tile pair is directly the A-matrix of the P@V WMMA.
//    Pipeline per chunk:  [issue V+mask loads] || score WMMAs ->
//    [issue next-K loads] || softmax VALU -> acc rescale -> P@V WMMAs.
// ---------------------------------------------------------------------------
__global__ void attn_kernel(const _Float16* __restrict__ Q,
                            const _Float16* __restrict__ K,
                            const _Float16* __restrict__ Vt,
                            const unsigned char* __restrict__ mask,
                            float* __restrict__ out) {
    const int lane = threadIdx.x & 31;
    const int wave = threadIdx.x >> 5;
    const int job  = blockIdx.x * (blockDim.x >> 5) + wave;  // 0..4095
    const int qt   = job & 127;        // query tile (16 rows)
    const int bh   = job >> 7;         // 0..31 (block waves share K/V stream)
    const int b    = bh >> 4;
    const int h    = bh & 15;
    const int hl   = lane >> 4;
    const int l16  = lane & 15;        // this lane's query within the tile

    const _Float16* Qb = Q  + (size_t)bh * SEQ * HEAD_DIM;
    const _Float16* Kb = K  + (size_t)bh * SEQ * HEAD_DIM;
    const _Float16* Vb = Vt + (size_t)bh * HEAD_DIM * SEQ;

    // Q as B-matrix (n = query = l16, k = d), hoisted out of the key loop.
    const _Float16* qp = Qb + (size_t)(qt * 16 + l16) * HEAD_DIM + 8 * hl;
    const v16h bq0 = ldtile(qp);        // d 0..31
    const v16h bq1 = ldtile(qp + 32);   // d 32..63

    // mask row for this lane's query (bool stored 1 byte/elem)
    const unsigned char* mrow =
        mask + ((size_t)b * SEQ + (qt * 16 + l16)) * SEQ;

    float m_run = -1.0e30f;   // running row max (finite sentinel)
    float l_run = 0.0f;       // running row sum
    v8f acc0 = {}, acc1 = {}, acc2 = {}, acc3 = {};

    // prologue: K A-tiles for chunk 0 (lane = key row, per-lane d split)
    v16h ka00 = ldtile(Kb + (size_t)(l16)      * HEAD_DIM + 8 * hl);
    v16h ka01 = ldtile(Kb + (size_t)(l16)      * HEAD_DIM + 8 * hl + 32);
    v16h ka10 = ldtile(Kb + (size_t)(16 + l16) * HEAD_DIM + 8 * hl);
    v16h ka11 = ldtile(Kb + (size_t)(16 + l16) * HEAD_DIM + 8 * hl + 32);

    for (int kk = 0; kk < SEQ; kk += 32) {
        // ---- issue V B-tile loads for this chunk (consumed at the bottom) ---
        const _Float16* vp = Vb + (size_t)l16 * SEQ + kk + 8 * hl;
        v16h vb0 = ldtile(vp + 0 * 16 * SEQ);
        v16h vb1 = ldtile(vp + 1 * 16 * SEQ);
        v16h vb2 = ldtile(vp + 2 * 16 * SEQ);
        v16h vb3 = ldtile(vp + 3 * 16 * SEQ);

        // mask bytes: keys kk + r + 8*hl (tile 0) / +16 (tile 1)
        const uint64_t mk0 = *(const uint64_t*)(mrow + kk + 8 * hl);
        const uint64_t mk1 = *(const uint64_t*)(mrow + kk + 16 + 8 * hl);

        SCHED_FENCE();   // keep V/mask loads issued before the score WMMAs

        // ---- S^T = K * Q^T  (Q pre-scaled by 1/sqrt(HD)) --------------------
        v8f s0 = {}, s1 = {};
        s0 = wmma_f16(ka00, bq0, s0);
        s0 = wmma_f16(ka01, bq1, s0);
        s1 = wmma_f16(ka10, bq0, s1);
        s1 = wmma_f16(ka11, bq1, s1);

        // ---- issue next chunk's K loads (hidden under softmax + P@V) --------
        const int kkn = (kk + 32 < SEQ) ? kk + 32 : kk;   // clamp: branch-free
        ka00 = ldtile(Kb + (size_t)(kkn + l16)      * HEAD_DIM + 8 * hl);
        ka01 = ldtile(Kb + (size_t)(kkn + l16)      * HEAD_DIM + 8 * hl + 32);
        ka10 = ldtile(Kb + (size_t)(kkn + 16 + l16) * HEAD_DIM + 8 * hl);
        ka11 = ldtile(Kb + (size_t)(kkn + 16 + l16) * HEAD_DIM + 8 * hl + 32);

        if (kk + 64 < SEQ) {   // L0/L2 prefetch two chunks ahead
            __builtin_prefetch(Kb + (size_t)(kk + 64 + l16) * HEAD_DIM, 0, 0);
            __builtin_prefetch(Vb + (size_t)l16 * SEQ + kk + 64, 0, 0);
        }

        SCHED_FENCE();   // pin next-K loads above the softmax VALU block

        // ---- mask + chunk max (rows are per-lane: lane = query) -------------
        float cmax = -1.0e30f;
#pragma unroll
        for (int r = 0; r < 8; ++r) {
            float v0 = s0[r];
            float v1 = s1[r];
            if ((mk0 >> (8 * r)) & 1ULL) v0 = -__builtin_inff();
            if ((mk1 >> (8 * r)) & 1ULL) v1 = -__builtin_inff();
            s0[r] = v0; s1[r] = v1;
            cmax = fmaxf(cmax, fmaxf(v0, v1));
        }
        cmax = fmaxf(cmax, __shfl_xor(cmax, 16, 32));  // merge lane-half pair
        const float m_new = fmaxf(m_run, cmax);
        const float scale = __expf(m_run - m_new);

        // p = exp(s - m_new); pack directly into A-matrix layout for P@V
        float csum = 0.0f;
        v16h ap;
#pragma unroll
        for (int r = 0; r < 8; ++r) {
            const float p0 = __expf(s0[r] - m_new);
            const float p1 = __expf(s1[r] - m_new);
            csum += p0 + p1;
            ap[r]     = (_Float16)p0;   // keys kk + r + 8*hl
            ap[8 + r] = (_Float16)p1;   // keys kk + 16 + r + 8*hl
        }
        csum += __shfl_xor(csum, 16, 32);
        l_run = l_run * scale + csum;
        m_run = m_new;

        // rescale output accumulators (rows = queries r + 8*hl; gather scale)
#pragma unroll
        for (int r = 0; r < 8; ++r) {
            const float sr = __shfl(scale, r + 8 * hl, 32);
            acc0[r] *= sr; acc1[r] *= sr; acc2[r] *= sr; acc3[r] *= sr;
        }

        // ---- P @ V : B = V^T rows (lane = d), k = local key -----------------
        acc0 = wmma_f16(ap, vb0, acc0);
        acc1 = wmma_f16(ap, vb1, acc1);
        acc2 = wmma_f16(ap, vb2, acc2);
        acc3 = wmma_f16(ap, vb3, acc3);
    }

    // finalize: out[b, s, h*64 + d] fp32
    float* orow = out + ((size_t)b * SEQ + qt * 16) * MODEL_DIM + h * HEAD_DIM;
#pragma unroll
    for (int r = 0; r < 8; ++r) {
        const float lr  = __shfl(l_run, r + 8 * hl, 32);
        const float inv = 1.0f / lr;
        float* prow = orow + (size_t)(r + 8 * hl) * MODEL_DIM;
        prow[0 * 16 + l16] = acc0[r] * inv;
        prow[1 * 16 + l16] = acc1[r] * inv;
        prow[2 * 16 + l16] = acc2[r] * inv;
        prow[3 * 16 + l16] = acc3[r] * inv;
    }
}

// ---------------------------------------------------------------------------
extern "C" void kernel_launch(void* const* d_in, const int* in_sizes, int n_in,
                              void* d_out, int out_size, void* d_ws, size_t ws_size,
                              hipStream_t stream) {
    const float* x  = (const float*)d_in[0];
    const unsigned char* mask = (const unsigned char*)d_in[1]; // jnp bool = 1B
    const float* Wk = (const float*)d_in[2];
    const float* bk = (const float*)d_in[3];
    const float* Wq = (const float*)d_in[4];
    const float* bq = (const float*)d_in[5];
    const float* Wv = (const float*)d_in[6];
    const float* bv = (const float*)d_in[7];

    const size_t NX = (size_t)BATCH * SEQ * MODEL_DIM;      // 4 Mi elements
    const size_t NW = (size_t)MODEL_DIM * MODEL_DIM;        // 1 Mi elements

    _Float16* xh  = (_Float16*)d_ws;
    _Float16* Wqh = xh  + NX;
    _Float16* Wkh = Wqh + NW;
    _Float16* Wvh = Wkh + NW;
    _Float16* Qh  = Wvh + NW;     // [B,H,S,HD] f16  (pre-scaled by 1/8)
    _Float16* Kh  = Qh  + NX;     // [B,H,S,HD] f16
    _Float16* Vth = Kh  + NX;     // [B,H,HD,S] f16 (transposed)

    cvt_f32_f16x4<<<(unsigned)(NX / 4 / 256), 256, 0, stream>>>(x,  xh,  (int)(NX / 4));
    cvt_f32_f16x4<<<(unsigned)(NW / 4 / 256), 256, 0, stream>>>(Wq, Wqh, (int)(NW / 4));
    cvt_f32_f16x4<<<(unsigned)(NW / 4 / 256), 256, 0, stream>>>(Wk, Wkh, (int)(NW / 4));
    cvt_f32_f16x4<<<(unsigned)(NW / 4 / 256), 256, 0, stream>>>(Wv, Wvh, (int)(NW / 4));

    // 4096 waves per matrix: 512 blocks x 8 waves; y selects Q/K/V
    qkv_kernel<<<dim3(512, 3), 256, 0, stream>>>(xh, Wqh, Wkh, Wvh,
                                                 bq, bk, bv, Qh, Kh, Vth);

    // 4096 jobs (2 batch * 16 heads * 128 query tiles): 512 blocks x 8 waves
    attn_kernel<<<512, 256, 0, stream>>>(Qh, Kh, Vth, mask, (float*)d_out);
}
